// RecursiveEncoder_26577257628366
// MI455X (gfx1250) — matrix operand order repaired and checked by reading, not compiled, not run on hardware
//
#include <hip/hip_runtime.h>

#define N_NODES 20000
#define N_EDGES 640000
#define FEAT    256
#define HID     128
#define NUM_SEM 57
#define CF_K    313   // FEAT + NUM_SEM
#define CF_KP   320   // padded to 10 * 32
#define CF_KT   10
#define EG_K    268   // 2*HID + 8 + 4
#define EG_KP   288   // padded to 9 * 32
#define EG_KT   9

typedef __attribute__((ext_vector_type(16))) __bf16 v16bf;
typedef __attribute__((ext_vector_type(8)))  float  v8f;

union Frag { v16bf f; uint4 q[2]; };

__device__ __forceinline__ unsigned f2bf1(float x) {
    union { float f; unsigned u; } v; v.f = x;
    return (v.u + 0x7FFFu + ((v.u >> 16) & 1u)) >> 16;   // round-to-nearest-even
}
__device__ __forceinline__ unsigned pack2bf(float lo, float hi) {
    return (f2bf1(lo) & 0xFFFFu) | (f2bf1(hi) << 16);
}

// ---------------------------------------------------------------------------
// Pack a KxHID fp32 weight matrix into WMMA-B fragment order, bf16.
// Layout: dword index = ((kt*8 + nt)*32 + lane)*8 + v
//   lane<16 : n = nt*16 + lane,      k pair = kt*32 + (v>>2)*16 + 2(v&3), +1
//   lane>=16: n = nt*16 + lane-16,   k pair = kt*32 + (v>>2)*16 + 8 + 2(v&3), +1
// One K-tile (kt) = 2048 dwords = 8 KB.
// ---------------------------------------------------------------------------
__global__ void pack_w_kernel(const float* __restrict__ W, unsigned* __restrict__ out,
                              int kreal, int kt_count) {
    int tid = blockIdx.x * blockDim.x + threadIdx.x;
    int total = kt_count * 2048;
    if (tid >= total) return;
    int v    = tid & 7;
    int lane = (tid >> 3) & 31;
    int nt   = (tid >> 8) & 7;
    int kt   = tid >> 11;
    int n    = nt * 16 + (lane & 15);
    int k0   = kt * 32 + ((v >> 2) * 16) + ((lane >> 4) * 8) + 2 * (v & 3);
    float a = (k0     < kreal) ? W[(size_t)k0 * HID + n]       : 0.f;
    float b = (k0 + 1 < kreal) ? W[(size_t)(k0 + 1) * HID + n] : 0.f;
    out[tid] = pack2bf(a, b);
}

// fp32 child_feats (N_NODES x CF_K) -> bf16 row-major padded (N_NODES x CF_KP)
__global__ void pack_cf_kernel(const float* __restrict__ cf, unsigned short* __restrict__ out) {
    int tid = blockIdx.x * blockDim.x + threadIdx.x;
    if (tid >= N_NODES * CF_KP) return;
    int row = tid / CF_KP, c = tid - row * CF_KP;
    float v = (c < CF_K) ? cf[(size_t)row * CF_K + c] : 0.f;
    out[tid] = (unsigned short)f2bf1(v);
}

// fp32 h (N_NODES x HID) -> packed bf16 dwords (N_NODES x HID/2)
__global__ void h_to_bf_kernel(const float* __restrict__ h, unsigned* __restrict__ hbf) {
    int tid = blockIdx.x * blockDim.x + threadIdx.x;
    if (tid >= N_NODES * (HID / 2)) return;
    hbf[tid] = pack2bf(h[2 * tid], h[2 * tid + 1]);
}

__global__ void zero_kernel(unsigned* __restrict__ p, int n) {
    int tid = blockIdx.x * blockDim.x + threadIdx.x;
    if (tid < n) p[tid] = 0u;
}

// ---------------------------------------------------------------------------
// Child MLP: h = relu(A_bf16 @ Wc + b) * exists.  One wave = 16 rows x 128 cols.
// Weights staged through LDS in 2-K-tile chunks (16 KB) shared by all 4 waves;
// B fragments then come from ds_load_b128 (cheap DScnt waits instead of
// draining LOADcnt against global latency before every WMMA).
// ---------------------------------------------------------------------------
__global__ void __launch_bounds__(128)
child_gemm_kernel(const unsigned short* __restrict__ A,
                  const unsigned* __restrict__ Wp,
                  const float* __restrict__ bias,
                  const float* __restrict__ exists,
                  float* __restrict__ h) {
    __shared__ unsigned bsmem[2 * 2048];        // 16 KB weight chunk
    int w = threadIdx.x >> 5, lane = threadIdx.x & 31;
    int t = blockIdx.x * 4 + w;                 // 16-row tile index
    bool valid = (t * 16 < N_NODES);
    int rowA  = valid ? (t * 16 + (lane & 15)) : 0;   // clamp for safe loads
    int halfk = (lane >> 4) * 8;

    v8f acc[8] = {};
    for (int kc = 0; kc < CF_KT; kc += 2) {     // CF_KT = 10, always full pairs
        __syncthreads();
        {
            const uint4* gsrc = (const uint4*)(Wp + (size_t)kc * 2048);
            uint4* ldst = (uint4*)bsmem;
            for (int i = threadIdx.x; i < 1024; i += 128)
                ldst[i] = gsrc[i];
        }
        __syncthreads();
#pragma unroll
        for (int dk = 0; dk < 2; ++dk) {
            int kt = kc + dk;
            Frag a;
            const unsigned short* ab = A + (size_t)rowA * CF_KP + kt * 32 + halfk;
            a.q[0] = *(const uint4*)(ab);
            a.q[1] = *(const uint4*)(ab + 16);
            Frag b[8];
#pragma unroll
            for (int nt = 0; nt < 8; ++nt) {
                const uint4* bp = (const uint4*)(bsmem + ((dk * 8 + nt) * 32 + lane) * 8);
                b[nt].q[0] = bp[0];
                b[nt].q[1] = bp[1];
            }
#pragma unroll
            for (int nt = 0; nt < 8; ++nt)
                acc[nt] = __builtin_amdgcn_wmma_f32_16x16x32_bf16(
                    false, a.f, false, b[nt].f, (short)0, acc[nt], false, false);
        }
    }
    if (!valid) return;
    int rbase = (lane >> 4) * 8;
    for (int nt = 0; nt < 8; ++nt) {
        int col = nt * 16 + (lane & 15);
        float bb = bias[col];
        for (int i = 0; i < 8; ++i) {
            int node = t * 16 + rbase + i;
            float v = fmaxf(acc[nt][i] + bb, 0.f) * exists[node];
            h[(size_t)node * HID + col] = v;
        }
    }
}

// ---------------------------------------------------------------------------
// Edge pass: msg = relu([h[src] | h[dst] | ef] @ W + b); scatter-max by src.
// Block = 4 waves x 16 edges. Per-wave private LDS A tile (16 x 288 bf16) +
// shared 2-K-tile LDS weight chunk. Total LDS ~53.5 KB.
// ---------------------------------------------------------------------------
__global__ void __launch_bounds__(128)
edge_gemm_kernel(const unsigned* __restrict__ hbf,
                 const int* __restrict__ eidx,
                 const float* __restrict__ eto,
                 const float* __restrict__ efe,
                 const unsigned* __restrict__ Wp,
                 const float* __restrict__ bias,
                 unsigned* __restrict__ hnext) {
    __shared__ unsigned short smem[4][16][EG_KP]; // 36.9 KB A tiles
    __shared__ unsigned bsmem[2 * 2048];          // 16 KB weight chunk
    __shared__ int ssrc[4][16];
    int w = threadIdx.x >> 5, lane = threadIdx.x & 31;
    int e0 = blockIdx.x * 64 + w * 16;

    // Stage 16 edge rows: h[src](128) | h[dst](128) | ef(12, zero-pad to 32), bf16
#pragma unroll 4
    for (int r = 0; r < 16; ++r) {
        int e = e0 + r;
        int src = eidx[2 * e], dst = eidx[2 * e + 1];
        if (lane == 0) ssrc[w][r] = src;
        unsigned* row = (unsigned*)&smem[w][r][0];
        const unsigned* hs = hbf + (size_t)src * (HID / 2);
        const unsigned* hd = hbf + (size_t)dst * (HID / 2);
        row[2 * lane]          = hs[2 * lane];
        row[2 * lane + 1]      = hs[2 * lane + 1];
        row[64 + 2 * lane]     = hd[2 * lane];
        row[64 + 2 * lane + 1] = hd[2 * lane + 1];
        if (lane < 16) {
            int c0 = 2 * lane;
            float f0 = (c0 < 4) ? eto[(size_t)e * 4 + c0]
                     : (c0 < 12 ? efe[(size_t)e * 8 + c0 - 4] : 0.f);
            int c1 = c0 + 1;
            float f1 = (c1 < 4) ? eto[(size_t)e * 4 + c1]
                     : (c1 < 12 ? efe[(size_t)e * 8 + c1 - 4] : 0.f);
            row[128 + lane] = pack2bf(f0, f1);
        }
    }

    v8f acc[8] = {};
    int rowA  = lane & 15;
    int halfk = (lane >> 4) * 8;
    for (int kc = 0; kc < EG_KT; kc += 2) {       // EG_KT = 9: 4 pairs + 1 tail
        int nk = (EG_KT - kc >= 2) ? 2 : 1;
        __syncthreads();
        {
            const uint4* gsrc = (const uint4*)(Wp + (size_t)kc * 2048);
            uint4* ldst = (uint4*)bsmem;
            int n4 = nk * 512;
            for (int i = threadIdx.x; i < n4; i += 128)
                ldst[i] = gsrc[i];
        }
        __syncthreads();
        for (int dk = 0; dk < nk; ++dk) {
            int kt = kc + dk;
            Frag a;
            const unsigned short* ab = &smem[w][rowA][kt * 32 + halfk];
            a.q[0] = *(const uint4*)(ab);
            a.q[1] = *(const uint4*)(ab + 16);
            Frag b[8];
#pragma unroll
            for (int nt = 0; nt < 8; ++nt) {
                const uint4* bp = (const uint4*)(bsmem + ((dk * 8 + nt) * 32 + lane) * 8);
                b[nt].q[0] = bp[0];
                b[nt].q[1] = bp[1];
            }
#pragma unroll
            for (int nt = 0; nt < 8; ++nt)
                acc[nt] = __builtin_amdgcn_wmma_f32_16x16x32_bf16(
                    false, a.f, false, b[nt].f, (short)0, acc[nt], false, false);
        }
    }

    // Epilogue: bias + relu + segment-max scatter (nonneg -> uint atomicMax ok)
    int rbase = (lane >> 4) * 8;
    for (int nt = 0; nt < 8; ++nt) {
        int col = nt * 16 + (lane & 15);
        float bb = bias[col];
        for (int i = 0; i < 8; ++i) {
            float v = fmaxf(acc[nt][i] + bb, 0.f);
            int src = ssrc[w][rbase + i];
            atomicMax(hnext + (size_t)src * HID + col, __float_as_uint(v));
        }
    }
}

// Per-column max over all nodes (h >= 0 everywhere, so identity 0 is correct)
__global__ void nodemax_kernel(const float* __restrict__ h, float* __restrict__ parent, int off) {
    __shared__ float red[256];
    int col = blockIdx.x;
    float m = 0.f;
    for (int n = threadIdx.x; n < N_NODES; n += 256)
        m = fmaxf(m, h[(size_t)n * HID + col]);
    red[threadIdx.x] = m;
    __syncthreads();
    for (int s = 128; s > 0; s >>= 1) {
        if (threadIdx.x < s) red[threadIdx.x] = fmaxf(red[threadIdx.x], red[threadIdx.x + s]);
        __syncthreads();
    }
    if (threadIdx.x == 0) parent[off + col] = red[0];
}

// out = relu(parent(1x384) @ W_parent(384x256) + b)
__global__ void final_kernel(const float* __restrict__ parent, const float* __restrict__ Wp,
                             const float* __restrict__ bp, float* __restrict__ out) {
    int j = threadIdx.x;
    float s = bp[j];
    for (int k = 0; k < 3 * HID; ++k) s += parent[k] * Wp[(size_t)k * FEAT + j];
    out[j] = fmaxf(s, 0.f);
}

extern "C" void kernel_launch(void* const* d_in, const int* in_sizes, int n_in,
                              void* d_out, int out_size, void* d_ws, size_t ws_size,
                              hipStream_t stream) {
    const float* child_feats  = (const float*)d_in[0];
    const float* child_exists = (const float*)d_in[1];
    const float* eto          = (const float*)d_in[2];
    const float* efe          = (const float*)d_in[3];
    const int*   eidx         = (const int*)  d_in[4];
    const float* W_child      = (const float*)d_in[5];
    const float* b_child      = (const float*)d_in[6];
    const float* W_e0         = (const float*)d_in[7];
    const float* b_e0         = (const float*)d_in[8];
    const float* W_e1         = (const float*)d_in[9];
    const float* b_e1         = (const float*)d_in[10];
    const float* W_par        = (const float*)d_in[11];
    const float* b_par        = (const float*)d_in[12];

    char* ws = (char*)d_ws;
    size_t off = 0;
    auto alloc = [&](size_t bytes) {
        void* p = ws + off;
        off += (bytes + 255) & ~(size_t)255;
        return p;
    };
    unsigned*       WcP  = (unsigned*)      alloc((size_t)CF_KT * 2048 * 4);
    unsigned*       We0P = (unsigned*)      alloc((size_t)EG_KT * 2048 * 4);
    unsigned*       We1P = (unsigned*)      alloc((size_t)EG_KT * 2048 * 4);
    unsigned short* Acf  = (unsigned short*)alloc((size_t)N_NODES * CF_KP * 2);
    float*          hA   = (float*)         alloc((size_t)N_NODES * HID * 4);
    float*          hB   = (float*)         alloc((size_t)N_NODES * HID * 4);
    unsigned*       hbf  = (unsigned*)      alloc((size_t)N_NODES * (HID / 2) * 4);
    float*          par  = (float*)         alloc(3 * HID * 4);

    // Weight / activation packing (bf16, WMMA fragment order)
    pack_w_kernel<<<CF_KT * 8, 256, 0, stream>>>(W_child, WcP, CF_K, CF_KT);
    pack_w_kernel<<<EG_KT * 8, 256, 0, stream>>>(W_e0, We0P, EG_K, EG_KT);
    pack_w_kernel<<<EG_KT * 8, 256, 0, stream>>>(W_e1, We1P, EG_K, EG_KT);
    pack_cf_kernel<<<(N_NODES * CF_KP + 255) / 256, 256, 0, stream>>>(child_feats, Acf);

    // Stage 1: child MLP
    child_gemm_kernel<<<(N_NODES / 16 + 3) / 4, 128, 0, stream>>>(Acf, WcP, b_child, child_exists, hA);
    nodemax_kernel<<<HID, 256, 0, stream>>>(hA, par, 0);

    // Stage 2: edge iteration 0
    h_to_bf_kernel<<<(N_NODES * (HID / 2) + 255) / 256, 256, 0, stream>>>(hA, hbf);
    zero_kernel<<<(N_NODES * HID + 255) / 256, 256, 0, stream>>>((unsigned*)hB, N_NODES * HID);
    edge_gemm_kernel<<<N_EDGES / 64, 128, 0, stream>>>(hbf, eidx, eto, efe, We0P, b_e0, (unsigned*)hB);
    nodemax_kernel<<<HID, 256, 0, stream>>>(hB, par, HID);

    // Stage 3: edge iteration 1
    h_to_bf_kernel<<<(N_NODES * (HID / 2) + 255) / 256, 256, 0, stream>>>(hB, hbf);
    zero_kernel<<<(N_NODES * HID + 255) / 256, 256, 0, stream>>>((unsigned*)hA, N_NODES * HID);
    edge_gemm_kernel<<<N_EDGES / 64, 128, 0, stream>>>(hbf, eidx, eto, efe, We1P, b_e1, (unsigned*)hA);
    nodemax_kernel<<<HID, 256, 0, stream>>>(hA, par, 2 * HID);

    // Stage 4: parent head
    final_kernel<<<1, 256, 0, stream>>>(par, W_par, b_par, (float*)d_out);
}
